// CSPNet_10230612099860
// MI455X (gfx1250) — matrix operand order repaired and checked by reading, not compiled, not run on hardware
//
#include <hip/hip_runtime.h>

// ---- problem constants (fixed by the reference) ----
#define G_    1024
#define NA_   20
#define H_    128
#define LAT_  256
#define L_    4
#define CHROW 80      // edge rows per chunk (4 source nodes x 20 dst)
#define NCHNK 5       // 5 * 80 = 400 edges per graph

// precomputed-weight workspace layout (in bf16 halves)
#define WS_EMBED_BLK   16384            // [128][128]
#define WS_LAYER_BASE  (3 * WS_EMBED_BLK)
#define WS_W1A   0
#define WS_W1B   16384
#define WS_W1D   32768                  // [128][64]
#define WS_W2    40960
#define WS_NW1T  57344
#define WS_NW1B  73728
#define WS_NW2   90112
#define WS_LAYER_SZ 106496
#define WS_TOTAL_HALVES (WS_LAYER_BASE + L_ * WS_LAYER_SZ)  // 475136

// ---- Tensor Data Mover availability (gfx1250 device pass only) ----
#if defined(__HIP_DEVICE_COMPILE__) && defined(__gfx1250__)
#  if __has_builtin(__builtin_amdgcn_tensor_load_to_lds)
#    define USE_TDM 1
#  else
#    define USE_TDM 0
#  endif
#else
#  define USE_TDM 0
#endif

typedef __attribute__((ext_vector_type(16))) __bf16 v16bf;
typedef __attribute__((ext_vector_type(8)))  float  v8f;
typedef unsigned int v4u __attribute__((ext_vector_type(4)));
typedef int          v8i __attribute__((ext_vector_type(8)));
typedef int          v4i __attribute__((ext_vector_type(4)));

__device__ __forceinline__ v8f vzero8() {
  v8f z;
#pragma unroll
  for (int i = 0; i < 8; ++i) z[i] = 0.f;
  return z;
}

__device__ __forceinline__ v8f bfmma(v16bf a, v16bf b, v8f c) {
  // D = A(16x32 bf16) * B(32x16 bf16) + C(16x16 f32)
  return __builtin_amdgcn_wmma_f32_16x16x32_bf16(false, a, false, b, (short)0, c,
                                                 false, false);
}

#if USE_TDM
// 1-D global->LDS DMA via the Tensor Data Mover (D# per CDNA5 ISA ch.8).
// Expressed as a 2D tile of `nbytes/8` 8-byte elements x 1 row.
__device__ __forceinline__ void tdm_load_1d(unsigned lds_off, const void* gsrc,
                                            unsigned nbytes) {
  const unsigned long long ga = (unsigned long long)(size_t)gsrc;
  const unsigned n8 = nbytes >> 3;  // 8-byte units (data_size code 3)

  v4u g0;
  g0[0] = 1u;                                   // count=1, user mode, no gather
  g0[1] = lds_off;                              // lds_addr (bytes)
  g0[2] = (unsigned)(ga & 0xffffffffu);         // global_addr[31:0]
  g0[3] = (unsigned)((ga >> 32) & 0x01ffffffu)  // global_addr[56:32]
          | 0x80000000u;                        // type=2 ("image")

  v8i g1;
  g1[0] = (int)(3u << 16);                // wg_mask=0, data_size=3 (8B), no opts
  g1[1] = (int)((n8 & 0xffffu) << 16);    // tensor_dim0[15:0] (bits 63:48)
  g1[2] = (int)((n8 >> 16) | (1u << 16)); // tensor_dim0[31:16] | tensor_dim1=1
  g1[3] = (int)((n8 & 0xffffu) << 16);    // tile_dim0 = n8 (bits 127:112)
  g1[4] = 1;                              // tile_dim1 = 1 (tile_dim2 = 0)
  g1[5] = (int)n8;                        // tensor_dim0_stride[31:0]
  g1[6] = (int)((n8 & 0xffff0000u));      // stride0[47:32]=0 | stride1[15:0]... = n8 hi16
  g1[6] = (int)(((n8)&0xffffu) << 16);    // tensor_dim1_stride[15:0] = n8 low
  g1[7] = (int)(n8 >> 16);                // tensor_dim1_stride[47:16] high part

  v4i gz;  gz[0] = 0; gz[1] = 0; gz[2] = 0; gz[3] = 0;   // groups 2/3 unused (2D)
#if defined(__clang_major__) && __clang_major__ >= 23
  v8i gz8; for (int i = 0; i < 8; ++i) gz8[i] = 0;
  __builtin_amdgcn_tensor_load_to_lds(g0, g1, gz, gz, gz8, 0);
#else
  __builtin_amdgcn_tensor_load_to_lds(g0, g1, gz, gz, 0);
#endif
}
#endif  // USE_TDM

// A fragment (16x32 bf16) from LDS, row-major with row stride `srow` halves.
// ISA layout: lanes 0-15 row M=lane, halves {K0..7, K16..23}; lanes 16-31 same
// row, halves {K8..15, K24..31}  -> two contiguous 16B LDS loads per lane.
__device__ __forceinline__ v16bf load_frag_A(const __bf16* base, int row0,
                                             int srow, int k0) {
  const int lane = threadIdx.x & 31;
  const __bf16* p = base + (row0 + (lane & 15)) * srow + k0 + (lane >> 4) * 8;
  v16bf a;
  ((uint4*)&a)[0] = *(const uint4*)p;         // K = k0 + {0..7}/{8..15}
  ((uint4*)&a)[1] = *(const uint4*)(p + 16);  // K = k0 + 16 + {0..7}/{8..15}
  return a;
}

// B fragment (32x16 bf16) from LDS weights stored transposed as [N][Kpad].
// lanes 0-15: col N=lane, K = k0+{0..7,16..23}; lanes 16-31: same col, K = k0+{8..15,24..31}
__device__ __forceinline__ v16bf load_frag_B(const __bf16* wT, int n0,
                                             int skcol, int k0) {
  const int lane = threadIdx.x & 31;
  const __bf16* p = wT + (n0 + (lane & 15)) * skcol + k0 + (lane >> 4) * 8;
  v16bf b;
  ((uint4*)&b)[0] = *(const uint4*)p;
  ((uint4*)&b)[1] = *(const uint4*)(p + 16);
  return b;
}

// Stage one weight block into LDS. Fast path: TDM DMA (or b128 copy) of the
// pre-converted transposed bf16 image. Fallback: fp32 transpose + convert.
__device__ __forceinline__ void stage_block(__bf16* dst, const __bf16* pre,
                                            const float* W, int K, int kpad) {
  if (pre) {
#if USE_TDM
    if ((threadIdx.x >> 5) == 0) {  // one wave issues the DMA descriptor
      tdm_load_1d((unsigned)(size_t)dst, pre, (unsigned)(128 * kpad * 2));
      __builtin_amdgcn_s_wait_tensorcnt((short)0);
    }
    // caller's __syncthreads() publishes the LDS tile to all waves
#else
    const int cnt = (128 * kpad) >> 3;  // uint4 = 8 halves
    const uint4* s = (const uint4*)pre;
    uint4* d = (uint4*)dst;
    for (int i = threadIdx.x; i < cnt; i += 256) d[i] = s[i];
#endif
  } else {
    for (int idx = threadIdx.x; idx < 128 * kpad; idx += 256) {
      const int n = idx / kpad, k = idx - n * kpad;
      dst[idx] = (__bf16)((k < K) ? W[k * 128 + n] : 0.f);
    }
  }
}

// 32xK @ Kx128 GEMM: wave `wid` owns columns [wid*16, wid*16+16), mtiles 0,1.
__device__ __forceinline__ void gemm_acc(const __bf16* A, int as,
                                         const __bf16* Wt, int ws, int K,
                                         v8f* acc) {
  const int nt = threadIdx.x >> 5;
  for (int k0 = 0; k0 < K; k0 += 32) {
    v16bf b  = load_frag_B(Wt, nt * 16, ws, k0);
    v16bf a0 = load_frag_A(A, 0, as, k0);
    acc[0] = bfmma(a0, b, acc[0]);
    v16bf a1 = load_frag_A(A, 16, as, k0);
    acc[1] = bfmma(a1, b, acc[1]);
  }
}

__device__ __forceinline__ float silu(float x) {
  return x / (1.f + __expf(-x));
}

// one-time fp32 -> transposed-bf16 weight conversion: dst[n*kpad+k] = W[k][n]
__global__ __launch_bounds__(256) void conv_wT(__bf16* __restrict__ dst,
                                               const float* __restrict__ W,
                                               int K, int kpad) {
  const int idx = blockIdx.x * 256 + threadIdx.x;
  if (idx < 128 * kpad) {
    const int n = idx / kpad, k = idx - n * kpad;
    dst[idx] = (__bf16)((k < K) ? W[k * 128 + n] : 0.f);
  }
}

__global__ __launch_bounds__(256) void cspnet_fused(
    const float* __restrict__ t, const int* __restrict__ atom_types,
    const float* __restrict__ frac_coords, const float* __restrict__ lattices,
    const float* __restrict__ node_emb, const float* __restrict__ atom_lat_w,
    const float* __restrict__ atom_lat_b, const float* __restrict__ edge_w1,
    const float* __restrict__ edge_b1, const float* __restrict__ edge_w2,
    const float* __restrict__ edge_b2, const float* __restrict__ node_w1,
    const float* __restrict__ node_b1, const float* __restrict__ node_w2,
    const float* __restrict__ node_b2, const float* __restrict__ coord_w,
    const float* __restrict__ lattice_w, const __bf16* __restrict__ wspre,
    float* __restrict__ out_lat, float* __restrict__ out_coord) {
  const int g    = blockIdx.x;
  const int tid  = threadIdx.x;
  const int lane = tid & 31;
  const int wid  = tid >> 5;
  const int nsub = lane & 15;
  const int mlo  = (lane >> 4) * 8;  // C layout: m = vgpr + mlo

  __shared__ __align__(16) float  sh_h[32 * H_];      // node features (f32 master)
  __shared__ __align__(16) float  sh_A[NA_ * H_];     // h@W1a + lat@W1c + b1 (per src)
  __shared__ __align__(16) float  sh_B[NA_ * H_];     // h@W1b (per dst)
  __shared__ __align__(16) float  sh_agg[NA_ * H_];   // scatter-sum target
  __shared__ __align__(16) float  sh_latc[H_];
  __shared__ __align__(16) float  sh_bias[H_];
  __shared__ __align__(16) float  sh_frac[64];
  __shared__ __align__(16) __bf16 sh_hbf[32 * H_];    // bf16 A operand / tmp
  __shared__ __align__(16) __bf16 sh_cbuf[CHROW * 64];// demb chunk / agg bf16
  __shared__ __align__(16) __bf16 sh_ef1[CHROW * H_]; // edge hidden (bf16)
  __shared__ __align__(16) __bf16 sh_wbuf[128 * 192]; // staged weights (transposed)

  if (tid < NA_ * 3) sh_frac[tid] = frac_coords[g * NA_ * 3 + tid];

  // ---------------- initial embedding: h = [emb | t] @ atom_lat_w + b -------
  {
    v8f acc[2] = {vzero8(), vzero8()};
    for (int s = 0; s < 3; ++s) {  // three K-chunks of 128 over K=384
      __syncthreads();
      for (int idx = tid; idx < 32 * H_; idx += 256) {
        const int r = idx >> 7, k = idx & 127;
        float v = 0.f;
        if (r < NA_)
          v = (s == 0) ? node_emb[atom_types[g * NA_ + r] * H_ + k]
                       : t[g * LAT_ + (s - 1) * 128 + k];
        sh_hbf[idx] = (__bf16)v;
      }
      stage_block(sh_wbuf, wspre ? wspre + s * WS_EMBED_BLK : nullptr,
                  atom_lat_w + s * 128 * H_, 128, 128);
      __syncthreads();
      gemm_acc(sh_hbf, H_, sh_wbuf, 128, 128, acc);
    }
    __syncthreads();
    const int n = wid * 16 + nsub;
#pragma unroll
    for (int mt = 0; mt < 2; ++mt)
#pragma unroll
      for (int v = 0; v < 8; ++v) {
        const int m = mt * 16 + mlo + v;
        if (m < NA_) sh_h[m * H_ + n] = acc[mt][v] + atom_lat_b[n];
      }
  }

  const float inv_deg = 1.f / (float)NA_;  // fully connected incl. self loops

  // --------------------------- message-passing layers -----------------------
  for (int layer = 0; layer < L_; ++layer) {
    const float*  W1  = edge_w1 + layer * 322 * H_;
    const float*  b1  = edge_b1 + layer * H_;
    const float*  W2  = edge_w2 + layer * H_ * H_;
    const float*  b2  = edge_b2 + layer * H_;
    const float*  nW1 = node_w1 + layer * 2 * H_ * H_;
    const float*  nb1 = node_b1 + layer * H_;
    const float*  nW2 = node_w2 + layer * H_ * H_;
    const float*  nb2 = node_b2 + layer * H_;
    const __bf16* pL =
        wspre ? wspre + WS_LAYER_BASE + layer * WS_LAYER_SZ : nullptr;

    __syncthreads();
    // h -> bf16 (rows >= NA_ zeroed)
    for (int idx = tid; idx < 32 * H_; idx += 256) {
      const int r = idx >> 7;
      sh_hbf[idx] = (__bf16)((r < NA_) ? sh_h[idx] : 0.f);
    }
    // latc = b1 + lattice @ W1c  (rows 256..261 of W1)
    if (tid < H_) {
      float s = b1[tid];
#pragma unroll
      for (int k = 0; k < 6; ++k)
        s += lattices[g * 6 + k] * W1[(256 + k) * H_ + tid];
      sh_latc[tid] = s;
    }
    if (tid >= H_ && tid < 2 * H_) sh_bias[tid - H_] = b2[tid - H_];
    for (int idx = tid; idx < NA_ * H_; idx += 256) sh_agg[idx] = 0.f;

    // ---- A term: h @ W1a (+ latc, +b1) ----
    stage_block(sh_wbuf, pL ? pL + WS_W1A : nullptr, W1, 128, 128);
    __syncthreads();
    {
      v8f acc[2] = {vzero8(), vzero8()};
      gemm_acc(sh_hbf, H_, sh_wbuf, 128, 128, acc);
      const int n = wid * 16 + nsub;
#pragma unroll
      for (int mt = 0; mt < 2; ++mt)
#pragma unroll
        for (int v = 0; v < 8; ++v) {
          const int m = mt * 16 + mlo + v;
          if (m < NA_) sh_A[m * H_ + n] = acc[mt][v] + sh_latc[n];
        }
    }
    __syncthreads();
    // ---- B term: h @ W1b ----
    stage_block(sh_wbuf, pL ? pL + WS_W1B : nullptr, W1 + 128 * H_, 128, 128);
    __syncthreads();
    {
      v8f acc[2] = {vzero8(), vzero8()};
      gemm_acc(sh_hbf, H_, sh_wbuf, 128, 128, acc);
      const int n = wid * 16 + nsub;
#pragma unroll
      for (int mt = 0; mt < 2; ++mt)
#pragma unroll
        for (int v = 0; v < 8; ++v) {
          const int m = mt * 16 + mlo + v;
          if (m < NA_) sh_B[m * H_ + n] = acc[mt][v];
        }
    }
    __syncthreads();
    // ---- stage W1d^T (K=60 pad 64) and W2^T (K=128) together ----
    stage_block(sh_wbuf, pL ? pL + WS_W1D : nullptr, W1 + 262 * H_, 60, 64);
    stage_block(sh_wbuf + 128 * 64, pL ? pL + WS_W2 : nullptr, W2, 128, 128);
    __syncthreads();

    // ---- edge chunks: demb -> GEMM1 -> silu -> GEMM2 -> silu -> scatter ----
    for (int c = 0; c < NCHNK; ++c) {
      for (int idx = tid; idx < CHROW * 30; idx += 256) {
        const int r = idx / 30, q = idx - r * 30;
        const int dim = q / 10, f = q - dim * 10;
        const int i = c * 4 + r / NA_, j = r % NA_;
        float d = sh_frac[j * 3 + dim] - sh_frac[i * 3 + dim];
        d -= floorf(d);  // mod 1
        const float ang = 6.28318530717958647692f * (float)f * d;
        sh_cbuf[r * 64 + q]      = (__bf16)__sinf(ang);
        sh_cbuf[r * 64 + 30 + q] = (__bf16)__cosf(ang);
      }
      for (int idx = tid; idx < CHROW * 4; idx += 256)
        sh_cbuf[(idx >> 2) * 64 + 60 + (idx & 3)] = (__bf16)0.f;
      __syncthreads();

      {  // GEMM1: demb(80x64) @ W1d + A[src] + B[dst] -> silu -> ef1
        const int nt = wid, n = nt * 16 + nsub;
        v16bf bq0 = load_frag_B(sh_wbuf, nt * 16, 64, 0);
        v16bf bq1 = load_frag_B(sh_wbuf, nt * 16, 64, 32);
        for (int mt = 0; mt < 5; ++mt) {
          v8f cacc = vzero8();
          cacc = bfmma(load_frag_A(sh_cbuf, mt * 16, 64, 0), bq0, cacc);
          cacc = bfmma(load_frag_A(sh_cbuf, mt * 16, 64, 32), bq1, cacc);
#pragma unroll
          for (int v = 0; v < 8; ++v) {
            const int r = mt * 16 + mlo + v;        // 0..79
            const int i = c * 4 + r / NA_, j = r % NA_;
            float x = cacc[v] + sh_A[i * H_ + n] + sh_B[j * H_ + n];
            sh_ef1[r * H_ + n] = (__bf16)silu(x);
          }
        }
      }
      __syncthreads();
      {  // GEMM2: ef1(80x128) @ W2 + b2 -> silu -> scatter-add into agg[src]
        const int nt = wid, n = nt * 16 + nsub;
        const __bf16* w2T = sh_wbuf + 128 * 64;
        v16bf bq0 = load_frag_B(w2T, nt * 16, 128, 0);
        v16bf bq1 = load_frag_B(w2T, nt * 16, 128, 32);
        v16bf bq2 = load_frag_B(w2T, nt * 16, 128, 64);
        v16bf bq3 = load_frag_B(w2T, nt * 16, 128, 96);
        for (int mt = 0; mt < 5; ++mt) {
          v8f cacc = vzero8();
          cacc = bfmma(load_frag_A(sh_ef1, mt * 16, H_, 0), bq0, cacc);
          cacc = bfmma(load_frag_A(sh_ef1, mt * 16, H_, 32), bq1, cacc);
          cacc = bfmma(load_frag_A(sh_ef1, mt * 16, H_, 64), bq2, cacc);
          cacc = bfmma(load_frag_A(sh_ef1, mt * 16, H_, 96), bq3, cacc);
          // run-length reduce: a lane's 8 rows cross a /20 group boundary at
          // most once -> <=2 ds_add_f32 instead of 8
          int   curi = c * 4 + (mt * 16 + mlo) / NA_;
          float run  = 0.f;
#pragma unroll
          for (int v = 0; v < 8; ++v) {
            const int r = mt * 16 + mlo + v;
            const int i = c * 4 + r / NA_;
            const float x = silu(cacc[v] + sh_bias[n]);
            if (i != curi) {
              atomicAdd(&sh_agg[curi * H_ + n], run);
              run  = 0.f;
              curi = i;
            }
            run += x;
          }
          atomicAdd(&sh_agg[curi * H_ + n], run);
        }
      }
      __syncthreads();
    }

    // ---- node MLP: h += silu(silu([h|agg/deg] @ nW1 + nb1) @ nW2 + nb2) ----
    __bf16* aggbf = sh_cbuf;  // reuse as [32][128]
    for (int idx = tid; idx < 32 * H_; idx += 256) {
      const int r = idx >> 7;
      aggbf[idx] = (__bf16)((r < NA_) ? sh_agg[idx] * inv_deg : 0.f);
    }
    if (tid < H_) sh_bias[tid] = nb1[tid];
    stage_block(sh_wbuf, pL ? pL + WS_NW1T : nullptr, nW1, 128, 128);
    __syncthreads();
    v8f nacc[2] = {vzero8(), vzero8()};
    gemm_acc(sh_hbf, H_, sh_wbuf, 128, 128, nacc);
    __syncthreads();
    stage_block(sh_wbuf, pL ? pL + WS_NW1B : nullptr, nW1 + 128 * H_, 128, 128);
    __syncthreads();
    gemm_acc(aggbf, H_, sh_wbuf, 128, 128, nacc);
    {  // tmp = silu(nacc + nb1) -> sh_hbf (h part of GEMM already consumed)
      const int n = wid * 16 + nsub;
#pragma unroll
      for (int mt = 0; mt < 2; ++mt)
#pragma unroll
        for (int v = 0; v < 8; ++v) {
          const int m = mt * 16 + mlo + v;
          const float x = silu(nacc[mt][v] + sh_bias[n]);
          sh_hbf[m * H_ + n] = (m < NA_) ? (__bf16)x : (__bf16)0.f;
        }
    }
    __syncthreads();
    if (tid < H_) sh_bias[tid] = nb2[tid];
    stage_block(sh_wbuf, pL ? pL + WS_NW2 : nullptr, nW2, 128, 128);
    __syncthreads();
    {
      v8f dacc[2] = {vzero8(), vzero8()};
      gemm_acc(sh_hbf, H_, sh_wbuf, 128, 128, dacc);
      const int n = wid * 16 + nsub;
#pragma unroll
      for (int mt = 0; mt < 2; ++mt)
#pragma unroll
        for (int v = 0; v < 8; ++v) {
          const int m = mt * 16 + mlo + v;
          if (m < NA_) sh_h[m * H_ + n] += silu(dacc[mt][v] + sh_bias[n]);
        }
    }
  }
  __syncthreads();

  // ------------------------------ outputs -----------------------------------
  for (int o = tid; o < NA_ * 3; o += 256) {  // coord_out = h @ coord_w
    const int r = o / 3, dch = o - r * 3;
    float s = 0.f;
    for (int k = 0; k < H_; ++k) s += sh_h[r * H_ + k] * coord_w[k * 3 + dch];
    out_coord[(g * NA_ + r) * 3 + dch] = s;
  }
  if (tid < H_) {  // graph mean
    float s = 0.f;
    for (int r = 0; r < NA_; ++r) s += sh_h[r * H_ + tid];
    sh_latc[tid] = s * inv_deg;
  }
  __syncthreads();
  if (tid < 6) {  // lattice_out = gfeat @ lattice_w
    float s = 0.f;
    for (int k = 0; k < H_; ++k) s += sh_latc[k] * lattice_w[k * 6 + tid];
    out_lat[g * 6 + tid] = s;
  }
}

extern "C" void kernel_launch(void* const* d_in, const int* in_sizes, int n_in,
                              void* d_out, int out_size, void* d_ws,
                              size_t ws_size, hipStream_t stream) {
  (void)in_sizes; (void)n_in; (void)out_size;
  const float* t        = (const float*)d_in[0];
  const int*   atypes   = (const int*)d_in[1];
  const float* frac     = (const float*)d_in[2];
  const float* latt     = (const float*)d_in[3];
  // d_in[4..6] = node2graph / edge_index / edge2graph: block-diagonal
  // fully-connected structure is implicit (edge k = src*20 + dst).
  const float* node_emb = (const float*)d_in[7];
  const float* alw      = (const float*)d_in[8];
  const float* alb      = (const float*)d_in[9];
  const float* ew1      = (const float*)d_in[10];
  const float* eb1      = (const float*)d_in[11];
  const float* ew2      = (const float*)d_in[12];
  const float* eb2      = (const float*)d_in[13];
  const float* nw1      = (const float*)d_in[14];
  const float* nb1      = (const float*)d_in[15];
  const float* nw2      = (const float*)d_in[16];
  const float* nb2      = (const float*)d_in[17];
  const float* cw       = (const float*)d_in[18];
  const float* lw       = (const float*)d_in[19];
  float* out = (float*)d_out;  // [G*6 lattice | N*3 coords]

  __bf16* pre = nullptr;
  if (ws_size >= (size_t)WS_TOTAL_HALVES * sizeof(__bf16)) {
    pre = (__bf16*)d_ws;
    for (int s = 0; s < 3; ++s)
      conv_wT<<<64, 256, 0, stream>>>(pre + s * WS_EMBED_BLK,
                                      alw + s * 128 * H_, 128, 128);
    for (int l = 0; l < L_; ++l) {
      __bf16* b = pre + WS_LAYER_BASE + l * WS_LAYER_SZ;
      const float* W1 = ew1 + l * 322 * H_;
      conv_wT<<<64, 256, 0, stream>>>(b + WS_W1A, W1, 128, 128);
      conv_wT<<<64, 256, 0, stream>>>(b + WS_W1B, W1 + 128 * H_, 128, 128);
      conv_wT<<<32, 256, 0, stream>>>(b + WS_W1D, W1 + 262 * H_, 60, 64);
      conv_wT<<<64, 256, 0, stream>>>(b + WS_W2, ew2 + l * H_ * H_, 128, 128);
      conv_wT<<<64, 256, 0, stream>>>(b + WS_NW1T, nw1 + l * 2 * H_ * H_, 128,
                                      128);
      conv_wT<<<64, 256, 0, stream>>>(b + WS_NW1B,
                                      nw1 + l * 2 * H_ * H_ + 128 * H_, 128,
                                      128);
      conv_wT<<<64, 256, 0, stream>>>(b + WS_NW2, nw2 + l * H_ * H_, 128, 128);
    }
  }

  cspnet_fused<<<G_, 256, 0, stream>>>(t, atypes, frac, latt, node_emb, alw,
                                       alb, ew1, eb1, ew2, eb2, nw1, nb1, nw2,
                                       nb2, cw, lw, pre, out, out + G_ * 6);
}
